// Spatial_Heterogeneity_AttentionLayer_43379169689894
// MI455X (gfx1250) — matrix operand order repaired and checked by reading, not compiled, not run on hardware
//
#include <hip/hip_runtime.h>
#include <hip/hip_bf16.h>

typedef __attribute__((ext_vector_type(16))) _Float16     v16h;
typedef __attribute__((ext_vector_type(8)))  _Float16     v8h;
typedef __attribute__((ext_vector_type(4)))  _Float16     v4h;
typedef __attribute__((ext_vector_type(8)))  float        v8f;
typedef __attribute__((ext_vector_type(4)))  unsigned int u32x4;
typedef __attribute__((ext_vector_type(8)))  int          i32x8;
typedef __attribute__((ext_vector_type(4)))  int          i32x4;

#define DIM   512
#define NNODE 307
#define NPAD  320          // 20 tiles of 16
#define PSTR  328          // LDS row stride: 328 % 64 = 8 -> no bank alias
#define ASTR  520          // proj A-tile LDS stride (halves): 520 % 64 = 8

#if defined(__gfx1250__) && __has_builtin(__builtin_amdgcn_tensor_load_to_lds) && \
    __has_builtin(__builtin_amdgcn_s_wait_tensorcnt)
#define USE_TDM 1
#else
#define USE_TDM 0
#endif

// ---------------------------------------------------------------- WMMA helpers

__device__ __forceinline__ v8f wmma_f16(v16h a, v16h b, v8f c) {
  return __builtin_amdgcn_wmma_f32_16x16x32_f16(false, a, false, b, (short)0, c,
                                                false, false);
}

__device__ __forceinline__ v8f vzero8() {
  v8f z;
#pragma unroll
  for (int i = 0; i < 8; ++i) z[i] = 0.0f;
  return z;
}

// A fragment (16x32, f16) from global f16 rows; row clamped to rowmax.
__device__ __forceinline__ v16h frag_a_g(const _Float16* __restrict__ base, int ldm,
                                         int row0, int rowmax, int k0) {
  const int lane = threadIdx.x & 31;
  int r = row0 + (lane & 15);
  if (r > rowmax) r = rowmax;
  const _Float16* p = base + (size_t)r * ldm + k0 + ((lane >> 4) << 3);
  v16h f;
#pragma unroll
  for (int i = 0; i < 8; ++i) { f[i] = p[i]; f[i + 8] = p[i + 16]; }
  return f;
}

// A fragment from LDS f16 (local rows 0..15 relative to P).
__device__ __forceinline__ v16h frag_a_lds(const _Float16* P, int ldm, int k0) {
  const int lane = threadIdx.x & 31;
  const _Float16* p = P + (lane & 15) * ldm + k0 + ((lane >> 4) << 3);
  v16h f;
#pragma unroll
  for (int i = 0; i < 8; ++i) { f[i] = p[i]; f[i + 8] = p[i + 16]; }
  return f;
}

// B fragment (32x16) where B[k][n] = src[n][k] (k contiguous per source row).
__device__ __forceinline__ v16h frag_b_rows(const _Float16* __restrict__ base, int ldm,
                                            int col0, int colmax, int k0) {
  const int lane = threadIdx.x & 31;
  int c = col0 + (lane & 15);
  if (c > colmax) c = colmax;
  const _Float16* p = base + (size_t)c * ldm + k0 + ((lane >> 4) << 3);
  v16h f;
#pragma unroll
  for (int i = 0; i < 8; ++i) { f[i] = p[i]; f[i + 8] = p[i + 16]; }
  return f;
}

// --------------------------------------------------------------- tile staging
// 32 rows x 512 cols into LDS with ASTR-half padded stride.

__device__ __forceinline__ void stage_tile(const float* __restrict__ Xrow,
                                           _Float16* Ash, int tid) {
  for (int i = tid; i < 32 * (DIM / 4); i += 128) {
    const int r = i >> 7, c4 = i & 127;
    const float4 x = *reinterpret_cast<const float4*>(Xrow + (size_t)r * DIM + c4 * 4);
    v4h h;
    h[0] = (_Float16)x.x; h[1] = (_Float16)x.y;
    h[2] = (_Float16)x.z; h[3] = (_Float16)x.w;
    *reinterpret_cast<v4h*>(&Ash[r * ASTR + c4 * 4]) = h;
  }
}

// f16 source: Tensor Data Mover copy when available (pad_interval=256 DWORDs,
// pad_amount=4 DWORDs reproduces the 520-half LDS stride), else vector copy.
__device__ __forceinline__ void stage_tile(const _Float16* __restrict__ Xrow,
                                           _Float16* Ash, int tid) {
#if USE_TDM
  if (tid < 32) {  // single wave issues the DMA (TDM ignores EXEC)
    u32x4 g0;
    const unsigned long long ga = (unsigned long long)(uintptr_t)Xrow;
    g0[0] = 1u;                                   // count=1, user descriptor
    g0[1] = (unsigned)(uintptr_t)Ash;             // lds_addr
    g0[2] = (unsigned)ga;                         // global_addr[31:0]
    g0[3] = (unsigned)((ga >> 32) & 0x1FFFFFFull) | (2u << 30);  // addr[56:32]|type=2
    i32x8 g1;
    g1[0] = (int)((1u << 16) |                    // data_size = 2 bytes
                  (1u << 20) |                    // pad_enable
                  (7u << 22) |                    // pad_interval = 256 DWORDs
                  (3u << 25));                    // pad_amount  = 4 DWORDs
    g1[1] = (int)(512u << 16);                    // tensor_dim0 = 512 (lo16)
    g1[2] = (int)((29472u & 0xffffu) << 16);      // tensor_dim1 (lo16)
    g1[3] = (int)(512u << 16);                    // tile_dim0 = 512
    g1[4] = 32;                                   // tile_dim1 = 32, tile_dim2 = 0
    g1[5] = 512;                                  // tensor_dim0_stride = 512
    g1[6] = 0;
    g1[7] = 0;
    i32x4 z4;
    z4[0] = z4[1] = z4[2] = z4[3] = 0;
#if __clang_major__ >= 23
    i32x8 z8;
#pragma unroll
    for (int i = 0; i < 8; ++i) z8[i] = 0;
    __builtin_amdgcn_tensor_load_to_lds(g0, g1, z4, z4, z8, 0);
#else
    __builtin_amdgcn_tensor_load_to_lds(g0, g1, z4, z4, 0);
#endif
    __builtin_amdgcn_s_wait_tensorcnt(0);
  }
#else
  for (int i = tid; i < 32 * (DIM / 8); i += 128) {
    const int r = i >> 6, c8 = i & 63;
    *reinterpret_cast<v8h*>(&Ash[r * ASTR + c8 * 8]) =
        *reinterpret_cast<const v8h*>(Xrow + (size_t)r * DIM + c8 * 8);
  }
#endif
}

// ---------------------------------------------------------------- mask softmax

__global__ __launch_bounds__(256)
void mask_softmax_kernel(const float* __restrict__ S, float* __restrict__ Mk) {
  __shared__ float red[256];
  const int row = blockIdx.x, tid = threadIdx.x;
  const float* r = S + (size_t)row * NNODE;
  float mx = -1e30f;
  for (int c = tid; c < NNODE; c += 256) mx = fmaxf(mx, r[c]);
  red[tid] = mx; __syncthreads();
  for (int s = 128; s; s >>= 1) { if (tid < s) red[tid] = fmaxf(red[tid], red[tid + s]); __syncthreads(); }
  mx = red[0]; __syncthreads();
  float sum = 0.0f;
  for (int c = tid; c < NNODE; c += 256) sum += __expf(r[c] - mx);
  red[tid] = sum; __syncthreads();
  for (int s = 128; s; s >>= 1) { if (tid < s) red[tid] += red[tid + s]; __syncthreads(); }
  const float inv = 1.0f / red[0];
  for (int c = tid; c < NNODE; c += 256)
    Mk[(size_t)row * NNODE + c] = __expf(r[c] - mx) * inv;
}

// --------------------------------------- weight transpose f32 -> f16 (4 at once)

__global__ __launch_bounds__(256)
void wtrans_kernel(const float* __restrict__ W0, const float* __restrict__ W1,
                   const float* __restrict__ W2, const float* __restrict__ W3,
                   _Float16* __restrict__ T0, _Float16* __restrict__ T1,
                   _Float16* __restrict__ T2, _Float16* __restrict__ T3) {
  const int i = blockIdx.x * 256 + threadIdx.x;
  if (i >= DIM * DIM) return;
  const int k = i >> 9, c = i & (DIM - 1);
  const float* W[4] = {W0, W1, W2, W3};
  _Float16*    T[4] = {T0, T1, T2, T3};
  const int w = blockIdx.y;
  T[w][(size_t)c * DIM + k] = (_Float16)W[w][i];
}

// ------------------------------- V transpose per head: Vt[bt][h][d][NPAD] (f16)

__global__ __launch_bounds__(256)
void vtrans_kernel(const _Float16* __restrict__ Vhw, _Float16* __restrict__ Vt) {
  const size_t i = (size_t)blockIdx.x * 256 + threadIdx.x;  // 96*512*NPAD
  const int    n    = (int)(i % NPAD);
  const size_t rowi = i / NPAD;                 // bt*512 + (h*64+d)
  const int    bt   = (int)(rowi >> 9);
  const int    d    = (int)(rowi & 511);
  _Float16 v = (_Float16)0.0f;
  if (n < NNODE) v = Vhw[((size_t)bt * NNODE + n) * DIM + d];
  Vt[i] = v;
}

// ------------------------------------------------------- projection GEMM (WMMA)
// Block: 32 rows x 512 cols, 4 waves x (2 row-tiles x 8 col-tiles) each.
// B fragments pipelined one deep with an unconditional wrap-around load
// (one dead 1 KB fragment per wave) -> no branches in the hot loop.

template <typename TIN, typename TOUT>
__global__ __launch_bounds__(128)
void proj_kernel(const TIN* __restrict__ X, const _Float16* __restrict__ Wt,
                 const float* __restrict__ bias, TOUT* __restrict__ Y) {
  __shared__ __align__(16) _Float16 Ash[32 * ASTR];
  const int m0   = blockIdx.x * 32;
  const int tid  = threadIdx.x;
  const int wave = tid >> 5;
  const int lane = tid & 31;

  stage_tile(X + (size_t)m0 * DIM, Ash, tid);
  __syncthreads();

  v8f acc0[8], acc1[8];
#pragma unroll
  for (int j = 0; j < 8; ++j) { acc0[j] = vzero8(); acc1[j] = vzero8(); }

  v16h b = frag_b_rows(Wt, DIM, (wave * 8) * 16, DIM - 1, 0);
  for (int ks = 0; ks < DIM; ks += 32) {
    const v16h a0 = frag_a_lds(Ash, ASTR, ks);
    const v16h a1 = frag_a_lds(Ash + 16 * ASTR, ASTR, ks);
#pragma unroll
    for (int j = 0; j < 8; ++j) {
      const int jn  = (j + 1) & 7;
      const int ksn = ((j == 7) ? ks + 32 : ks) & (DIM - 1);   // wrap: always valid
      const v16h bn = frag_b_rows(Wt, DIM, (wave * 8 + jn) * 16, DIM - 1, ksn);
      acc0[j] = wmma_f16(a0, b, acc0[j]);
      acc1[j] = wmma_f16(a1, b, acc1[j]);
      b = bn;
    }
  }

  const int coll  = lane & 15;
  const int rbase = (lane >> 4) * 8;
#pragma unroll
  for (int j = 0; j < 8; ++j) {
    const int c = (wave * 8 + j) * 16 + coll;
    const float bv = bias[c];
#pragma unroll
    for (int r = 0; r < 8; ++r) {
      Y[(size_t)(m0 + rbase + r) * DIM + c]      = (TOUT)(acc0[j][r] + bv);
      Y[(size_t)(m0 + 16 + rbase + r) * DIM + c] = (TOUT)(acc1[j][r] + bv);
    }
  }
}

// ----------------------------------------------------- fused attention (WMMA)
// One block: (b,h,t) x 32 query nodes (2 row-tiles share every K/V fragment).
// Phase 1 fully unrolled (5 tiles/wave) so all pipeline guards are constant.

__global__ __launch_bounds__(128)
void attn_kernel(const _Float16* __restrict__ Qh, const _Float16* __restrict__ Kh,
                 const _Float16* __restrict__ Vt, const float* __restrict__ Mk,
                 _Float16* __restrict__ Ah) {
  __shared__ __align__(16) float     Ssc[32 * PSTR];   // 41 KB
  __shared__ __align__(16) _Float16  Ph [32 * PSTR];   // 21 KB

  const int m0  = blockIdx.x * 32;
  const int bht = blockIdx.y;                 // B*H*T = 768
  const int t   = bht % 12;
  const int h   = (bht / 12) & 7;
  const int b   = bht / 96;
  const int bt  = b * 12 + t;
  const size_t base = ((size_t)bt * NNODE) * DIM + (size_t)h * 64;
  const _Float16* Qp  = Qh + base;
  const _Float16* Kp  = Kh + base;
  const _Float16* Vtp = Vt + ((size_t)bt * 8 + h) * 64 * NPAD;

  const int tid = threadIdx.x, wave = tid >> 5, lane = tid & 31;
  const bool rows_ok = (m0 + 31) < NNODE;     // uniform fast path (all but last block)

  // ---- phase 1: scores [32 x 320]; 5 tiles per wave, fully unrolled ----
  const v16h qa00 = frag_a_g(Qp, DIM, m0,      NNODE - 1, 0);
  const v16h qa01 = frag_a_g(Qp, DIM, m0,      NNODE - 1, 32);
  const v16h qa10 = frag_a_g(Qp, DIM, m0 + 16, NNODE - 1, 0);
  const v16h qa11 = frag_a_g(Qp, DIM, m0 + 16, NNODE - 1, 32);

  v16h kb0 = frag_b_rows(Kp, DIM, wave * 16, NNODE - 1, 0);
  v16h kb1 = frag_b_rows(Kp, DIM, wave * 16, NNODE - 1, 32);
  const int lr0 = (lane >> 4) * 8;

#pragma unroll
  for (int it = 0; it < NPAD / 64; ++it) {    // 5 iterations, ct = wave + 4*it
    const int ct = wave + 4 * it;
    const int c0 = ct * 16;
    if (it < 3)
      __builtin_prefetch(Kp + (size_t)(ct + 8) * 16 * DIM, 0, 0);  // global_prefetch_b8
    v16h nb0 = kb0, nb1 = kb1;
    if (it < 4) {                              // constant after unroll -> no branch
      nb0 = frag_b_rows(Kp, DIM, (ct + 4) * 16, NNODE - 1, 0);
      nb1 = frag_b_rows(Kp, DIM, (ct + 4) * 16, NNODE - 1, 32);
    }
    v8f s0 = vzero8(), s1 = vzero8();
    s0 = wmma_f16(qa00, kb0, s0);
    s0 = wmma_f16(qa01, kb1, s0);
    s1 = wmma_f16(qa10, kb0, s1);
    s1 = wmma_f16(qa11, kb1, s1);

    const int col  = c0 + (lane & 15);
    const int colC = col < NNODE ? col : NNODE - 1;
    if (rows_ok) {                       // constant-offset loads & stores
      const float* mp0 = Mk + (size_t)(m0 + lr0) * NNODE + colC;
      const float* mp1 = mp0 + (size_t)16 * NNODE;
      float* sp0 = Ssc + lr0 * PSTR + col;
      float* sp1 = sp0 + 16 * PSTR;
#pragma unroll
      for (int r = 0; r < 8; ++r) {
        sp0[r * PSTR] = s0[r] * 0.125f * mp0[(size_t)r * NNODE];
        sp1[r * PSTR] = s1[r] * 0.125f * mp1[(size_t)r * NNODE];
      }
    } else {
#pragma unroll
      for (int r = 0; r < 8; ++r) {
        int g0r = m0 + lr0 + r;      if (g0r > NNODE - 1) g0r = NNODE - 1;
        int g1r = m0 + 16 + lr0 + r; if (g1r > NNODE - 1) g1r = NNODE - 1;
        Ssc[(lr0 + r) * PSTR + col]      = s0[r] * 0.125f * Mk[(size_t)g0r * NNODE + colC];
        Ssc[(16 + lr0 + r) * PSTR + col] = s1[r] * 0.125f * Mk[(size_t)g1r * NNODE + colC];
      }
    }
    kb0 = nb0; kb1 = nb1;
  }
  __syncthreads();

  // ---- phase 2: row softmax over 307 valid cols; zero the pad cols of Ph ----
  {
    const int row8 = tid >> 3, sub = tid & 7;
    for (int row = row8; row < 32; row += 16) {
      float* rs = Ssc + row * PSTR;
      float mx = -1e30f;
      for (int c = sub; c < NNODE; c += 8) mx = fmaxf(mx, rs[c]);
#pragma unroll
      for (int off = 4; off; off >>= 1) mx = fmaxf(mx, __shfl_xor(mx, off, 8));
      float sum = 0.0f;
      for (int c = sub; c < NNODE; c += 8) {
        const float e = __expf(rs[c] - mx);
        rs[c] = e;
        sum += e;
      }
#pragma unroll
      for (int off = 4; off; off >>= 1) sum += __shfl_xor(sum, off, 8);
      const float inv = 1.0f / sum;
      for (int c = sub; c < NNODE; c += 8)
        Ph[row * PSTR + c] = (_Float16)(rs[c] * inv);
      for (int c = NNODE + sub; c < NPAD; c += 8)
        Ph[row * PSTR + c] = (_Float16)0.0f;   // pad cols contribute exactly 0
    }
  }
  __syncthreads();

  // ---- phase 3: out[32 x 64] = P[32 x 320] * V[320 x 64]; unrolled pipeline ----
  const int n0 = wave * 16;
  v8f o0 = vzero8(), o1 = vzero8();
  v16h vb = frag_b_rows(Vtp, NPAD, n0, 63, 0);
#pragma unroll
  for (int ks = 0; ks < NPAD; ks += 32) {      // 10 iterations, guards constant
    v16h vn = vb;
    if (ks + 32 < NPAD) vn = frag_b_rows(Vtp, NPAD, n0, 63, ks + 32);
    const v16h pa0 = frag_a_lds(Ph, PSTR, ks);
    const v16h pa1 = frag_a_lds(Ph + 16 * PSTR, PSTR, ks);
    o0 = wmma_f16(pa0, vb, o0);
    o1 = wmma_f16(pa1, vb, o1);
    vb = vn;
  }
  const int col = n0 + (lane & 15);
#pragma unroll
  for (int r = 0; r < 8; ++r) {
    const int g0r = m0 + lr0 + r;
    const int g1r = m0 + 16 + lr0 + r;
    if (g0r < NNODE) Ah[base + (size_t)g0r * DIM + col] = (_Float16)o0[r];
    if (g1r < NNODE) Ah[base + (size_t)g1r * DIM + col] = (_Float16)o1[r];
  }
}

// ---------------------------------------------------------------- host launch

extern "C" void kernel_launch(void* const* d_in, const int* in_sizes, int n_in,
                              void* d_out, int out_size, void* d_ws, size_t ws_size,
                              hipStream_t stream) {
  (void)in_sizes; (void)n_in; (void)out_size; (void)ws_size;
  const int B = 8, T = 12;
  const int M   = B * T * NNODE;    // 29472
  const int MT2 = M / 32;           // 921 row-blocks of 32

  const float* q   = (const float*)d_in[0];
  const float* k   = (const float*)d_in[1];
  const float* v   = (const float*)d_in[2];
  const float* sem = (const float*)d_in[3];
  const float* Wq  = (const float*)d_in[4];
  const float* bq  = (const float*)d_in[5];
  const float* Wk  = (const float*)d_in[6];
  const float* bk  = (const float*)d_in[7];
  const float* Wv  = (const float*)d_in[8];
  const float* bv  = (const float*)d_in[9];
  const float* Wo  = (const float*)d_in[10];
  const float* bo  = (const float*)d_in[11];

  char*  ws  = (char*)d_ws;
  size_t off = 0;
  auto carve = [&](size_t bytes) -> void* {
    void* p = ws + off;
    off += (bytes + 255) & ~(size_t)255;
    return p;
  };
  float*    maskws = (float*)carve((size_t)NNODE * NNODE * 4);
  _Float16* Wtq    = (_Float16*)carve((size_t)DIM * DIM * 2);
  _Float16* Wtk    = (_Float16*)carve((size_t)DIM * DIM * 2);
  _Float16* Wtv    = (_Float16*)carve((size_t)DIM * DIM * 2);
  _Float16* Wto    = (_Float16*)carve((size_t)DIM * DIM * 2);
  _Float16* Qhw    = (_Float16*)carve((size_t)M * DIM * 2);
  _Float16* Khw    = (_Float16*)carve((size_t)M * DIM * 2);
  _Float16* Vhw    = (_Float16*)carve((size_t)M * DIM * 2);
  _Float16* Vtw    = (_Float16*)carve((size_t)(B * T) * 8 * 64 * NPAD * 2);
  _Float16* Ahw    = (_Float16*)carve((size_t)M * DIM * 2);

  mask_softmax_kernel<<<NNODE, 256, 0, stream>>>(sem, maskws);

  wtrans_kernel<<<dim3((DIM * DIM + 255) / 256, 4), 256, 0, stream>>>(
      Wq, Wk, Wv, Wo, Wtq, Wtk, Wtv, Wto);

  proj_kernel<float, _Float16><<<MT2, 128, 0, stream>>>(q, Wtq, bq, Qhw);
  proj_kernel<float, _Float16><<<MT2, 128, 0, stream>>>(k, Wtk, bk, Khw);
  proj_kernel<float, _Float16><<<MT2, 128, 0, stream>>>(v, Wtv, bv, Vhw);

  {
    const size_t vt_elems = (size_t)(B * T) * 512 * NPAD;   // 96*512*320
    vtrans_kernel<<<(unsigned)((vt_elems + 255) / 256), 256, 0, stream>>>(Vhw, Vtw);
  }

  attn_kernel<<<dim3(NPAD / 32, B * 8 * T), 128, 0, stream>>>(Qhw, Khw, Vtw, maskws, Ahw);

  proj_kernel<_Float16, float><<<MT2, 128, 0, stream>>>(Ahw, Wto, bo, (float*)d_out);
}
